// GenePathwayTransformerEncoder_21242908246136
// MI455X (gfx1250) — compile-verified, matrix operand order checked by reading, not run on hardware
//
#include <hip/hip_runtime.h>
#include <hip/hip_bf16.h>
#include <stdint.h>

// ---------------------------------------------------------------------------
// Types for CDNA5 WMMA (wave32, 16x16x32 bf16 -> f32)
// ---------------------------------------------------------------------------
typedef __attribute__((ext_vector_type(16))) __bf16 v16bf;
typedef __attribute__((ext_vector_type(8)))  float  v8f;
typedef __attribute__((ext_vector_type(2)))  float  v2f;
typedef __attribute__((ext_vector_type(2)))  __bf16 v2bf;

union FragB { v16bf v; uint32_t u[8]; };
union AccF  { v8f   v; float    f[8]; };

__device__ __forceinline__ float bf2f(uint16_t u) {
    return __uint_as_float(((uint32_t)u) << 16);
}
// Pack two f32 into one dword of bf16 (low = a, high = b). Vector convert
// pattern-matches to a single two-operand v_cvt_pk_bf16_f32.
__device__ __forceinline__ uint32_t pack_bf16(float a, float b) {
    v2f t; t.x = a; t.y = b;
    v2bf r = __builtin_convertvector(t, v2bf);
    uint32_t u;
    __builtin_memcpy(&u, &r, sizeof(u));
    return u;
}
__device__ __forceinline__ void store_bf16(uint16_t* p, float x) {
    union { __bf16 h; uint16_t u; } r;
    r.h = (__bf16)x;
    *p = r.u;
}

// Load a 16(row) x 32(col) bf16 fragment from row-major memory with leading
// dimension ld (elements). Per ISA 7.12.2 (16-bit A-matrix 16x32):
//   lane: row = lane%16, half = lane/16 adds +8 to col-pair base
//   reg v: col pair K0 = 16*(v/4) + 2*(v%4) (+ 8*half)
// Same bit pattern serves the B operand when memory is (N x K) row-major.
// Works for global or LDS pointers; dword groups are 16B-aligned -> b128.
__device__ __forceinline__ v16bf load_frag(const uint16_t* __restrict__ base, int ld) {
    int lane = threadIdx.x & 31;
    const uint16_t* rowp = base + (size_t)(lane & 15) * ld + ((lane >> 4) << 3);
    FragB f;
#pragma unroll
    for (int v = 0; v < 8; ++v) {
        int K0 = ((v >> 2) << 4) + ((v & 3) << 1);
        f.u[v] = *(const uint32_t*)(rowp + K0);
    }
    return f.v;
}

__device__ __forceinline__ v8f wmma32(v16bf a, v16bf b, v8f c) {
    return __builtin_amdgcn_wmma_f32_16x16x32_bf16(false, a, false, b, (short)0, c,
                                                   false, false);
}

// ---------------------------------------------------------------------------
// Problem constants
// ---------------------------------------------------------------------------
#define NB   16
#define NG   2000
#define NP   512
#define ND   128
#define NH   4
#define DH   32
#define GT_TILES 125           // 2000/16

// LDS staging for attention (padded so the 16 fragment rows hit 16 distinct
// banks: K stride 40 elems = 20 dwords, V^T stride 520 elems = 260 dwords;
// 20*r mod 64 and 260*r mod 64 are distinct for r=0..15; both keep 16B align)
#define SK_LD   40
#define SVT_LD  520
#define SK_ELEMS  (NP * SK_LD)      // 20480
#define SVT_ELEMS (DH * SVT_LD)     // 16640
#define ATTN_LDS_BYTES ((SK_ELEMS + SVT_ELEMS) * 2)   // 74240

// ---------------------------------------------------------------------------
// Stage 0: gathers, sigmoid mask, bf16 staging, weight transposes
// ---------------------------------------------------------------------------
__global__ void mask_kernel(const int* __restrict__ gi, const float* __restrict__ Wsoft,
                            float* __restrict__ mask) {
    int i = blockIdx.x * blockDim.x + threadIdx.x;
    if (i >= NG * NP) return;
    int g = i >> 9, p = i & 511;
    int row = gi[NG + g];                      // reference uses gene_indices[1]
    float w = Wsoft[(size_t)row * NP + p];
    mask[i] = 1.0f / (1.0f + __expf(-w));
}

__global__ void xinit_kernel(const int* __restrict__ gi, const float* __restrict__ expr,
                             const float* __restrict__ gemb, uint16_t* __restrict__ xb) {
    int i = blockIdx.x * blockDim.x + threadIdx.x;
    if (i >= NB * NG * ND) return;
    int d = i & 127, bg = i >> 7;
    int idx = gi[bg];
    store_bf16(xb + i, gemb[(size_t)idx * ND + d] * expr[bg]);
}

__global__ void kxinit_kernel(const int* __restrict__ pi, const float* __restrict__ pemb,
                              uint16_t* __restrict__ kxb) {
    int i = blockIdx.x * blockDim.x + threadIdx.x;
    if (i >= NB * NP * ND) return;
    int d = i & 127, bp = i >> 7;
    int idx = pi[bp];
    store_bf16(kxb + i, pemb[(size_t)idx * ND + d]);
}

// src: (K x N) f32 row-major -> dst: (N x K) bf16 row-major
__global__ void transpose_bf16_kernel(const float* __restrict__ src,
                                      uint16_t* __restrict__ dst, int K, int N) {
    int i = blockIdx.x * blockDim.x + threadIdx.x;
    if (i >= K * N) return;
    int k = i / N, n = i % N;
    store_bf16(dst + (size_t)n * K + k, src[i]);
}

// ---------------------------------------------------------------------------
// Generic projection: Y = X @ W + b with X (rows x 128) bf16, WT (128 x 128)
// bf16 stored transposed (out,in). One wave per 16x16 output tile.
// transposed==0: Y row-major (rows x 128) bf16. Computed as Y^T = W^T x X^T
//   (A = WT tile, B = X tile) so C-regs are consecutive OUT columns of one
//   row -> reg-pair packed dword stores.
// transposed==1: Y is V^T, laid out (B, 128, 512): Y[b][d][p]. Computed as
//   Y = X x W (A = X tile, B = WT tile) so C-regs are consecutive rows (p)
//   -> packed dword stores into the transposed buffer.
// ---------------------------------------------------------------------------
__global__ void proj_kernel(const uint16_t* __restrict__ X, const uint16_t* __restrict__ WT,
                            const float* __restrict__ bias, uint16_t* __restrict__ Y,
                            int row_tiles, int transposed) {
    int job = (int)((blockIdx.x * blockDim.x + threadIdx.x) >> 5);
    if (job >= row_tiles * 8) return;
    int nt = job & 7, rt = job >> 3;
    int lane = threadIdx.x & 31;
    int half = lane >> 4, col = lane & 15;

    const uint16_t* Xb = X + (size_t)rt * 16 * ND;
    const uint16_t* Wb = WT + (size_t)nt * 16 * ND;
    AccF acc;

    if (!transposed) {
        // bias indexed by out-col = nt*16 + 8*half + r -> two float4 loads
        float4 b0 = *(const float4*)(bias + nt * 16 + 8 * half);
        float4 b1 = *(const float4*)(bias + nt * 16 + 8 * half + 4);
        acc.f[0] = b0.x; acc.f[1] = b0.y; acc.f[2] = b0.z; acc.f[3] = b0.w;
        acc.f[4] = b1.x; acc.f[5] = b1.y; acc.f[6] = b1.z; acc.f[7] = b1.w;
#pragma unroll
        for (int kc = 0; kc < 4; ++kc) {
            v16bf a = load_frag(Wb + kc * 32, ND);   // A = W^T rows (out, k)
            v16bf b = load_frag(Xb + kc * 32, ND);   // B = X^T from X rows
            acc.v = wmma32(a, b, acc.v);
        }
        // C: m = out-col (regs), n = row (lane) -> packed stores along out dim
        uint16_t* yrow = Y + (size_t)(rt * 16 + col) * ND + nt * 16 + 8 * half;
#pragma unroll
        for (int r = 0; r < 8; r += 2)
            *(uint32_t*)(yrow + r) = pack_bf16(acc.f[r], acc.f[r + 1]);
    } else {
        float bv = bias[nt * 16 + col];              // n = lane%16 for all regs
#pragma unroll
        for (int r = 0; r < 8; ++r) acc.f[r] = bv;
#pragma unroll
        for (int kc = 0; kc < 4; ++kc) {
            v16bf a = load_frag(Xb + kc * 32, ND);
            v16bf b = load_frag(Wb + kc * 32, ND);
            acc.v = wmma32(a, b, acc.v);
        }
#pragma unroll
        for (int r = 0; r < 8; r += 2) {
            int m = rt * 16 + r + 8 * half;          // global row over B*P
            int bidx = m >> 9, p = m & 511;          // tile never splits batches
            uint32_t pk = pack_bf16(acc.f[r], acc.f[r + 1]);
            *(uint32_t*)(Y + ((size_t)bidx * ND * NP + (size_t)(nt * 16 + col) * NP + p)) = pk;
        }
    }
}

// ---------------------------------------------------------------------------
// Fused masked-softmax attention. One block = 8 waves = one (b,h) pair and a
// chunk of 16 query tiles (each wave owns 2). The (b,h) slices of K (512x32)
// and V^T (32x512) are cooperatively staged into padded LDS and shared by all
// waves. Per wave: S^T = K x Q^T (softmax axis = regs + xor16), online flash
// softmax, in-lane repack of exp'd C-tiles into the next B operand,
// O^T += V^T x P^T.
// ---------------------------------------------------------------------------
extern __shared__ uint16_t attn_smem[];

__global__ __launch_bounds__(256) void
attn_kernel(const uint16_t* __restrict__ Qb, const uint16_t* __restrict__ Kb,
            const uint16_t* __restrict__ VTb, const float* __restrict__ mask,
            uint16_t* __restrict__ Ob) {
    uint16_t* sK  = attn_smem;                 // [512][SK_LD]  (first 32 cols used)
    uint16_t* sVT = attn_smem + SK_ELEMS;      // [32][SVT_LD]  (first 512 cols used)

    int tid   = threadIdx.x;
    int bh    = blockIdx.x >> 3;               // 0..63
    int chunk = blockIdx.x & 7;                // 0..7 -> 16 query tiles each
    int b = bh >> 2, h = bh & 3;

    // ---- cooperative LDS fill (b128 global loads -> b128 ds stores) ----
    {
        const uint16_t* kg = Kb + (size_t)b * NP * ND + h * DH;
        for (int idx = tid; idx < NP * 4; idx += 256) {     // 4 uint4 per K row
            int row = idx >> 2, q = idx & 3;
            *((uint4*)(sK + row * SK_LD) + q) =
                *((const uint4*)(kg + (size_t)row * ND) + q);
        }
        const uint16_t* vg = VTb + ((size_t)b * ND + h * DH) * NP;
        for (int idx = tid; idx < DH * 64; idx += 256) {    // 64 uint4 per V^T row
            int row = idx >> 6, q = idx & 63;
            *((uint4*)(sVT + row * SVT_LD) + q) =
                *((const uint4*)(vg + (size_t)row * NP) + q);
        }
    }
    __syncthreads();

    int w    = tid >> 5;
    int lane = tid & 31;
    int half = lane >> 4, col = lane & 15;     // col = query index within tile
    const float inv_sqrt = 0.17677669529663687f;  // 1/sqrt(dh=32)

    for (int sub = 0; sub < 2; ++sub) {
        int gt = chunk * 16 + w * 2 + sub;     // uniform per wave
        if (gt >= GT_TILES) break;
        int g = gt * 16 + col;

        v16bf qf = load_frag(Qb + ((size_t)(b * NG + gt * 16) * ND) + h * DH, ND);

        AccF acc0, acc1;
#pragma unroll
        for (int r = 0; r < 8; ++r) { acc0.f[r] = 0.0f; acc1.f[r] = 0.0f; }
        float mrun = -1e30f, lsum = 0.0f;
        const float* mrow = mask + (size_t)g * NP;

        for (int pc = 0; pc < 16; ++pc) {
            int p0 = pc * 32;
            v16bf k0 = load_frag(sK + (size_t)p0 * SK_LD, SK_LD);
            v16bf k1 = load_frag(sK + (size_t)(p0 + 16) * SK_LD, SK_LD);
            if (pc < 15)
                __builtin_prefetch(mrow + p0 + 32, 0, 1);

            v8f z = {};
            AccF s0, s1;
            s0.v = wmma32(k0, qf, z);
            s1.v = wmma32(k1, qf, z);

            // mask: S^T element (p = r+8*half, g = col) -> 8 consecutive floats
            float4 ma0 = *(const float4*)(mrow + p0 + 8 * half);
            float4 mb0 = *(const float4*)(mrow + p0 + 8 * half + 4);
            float4 ma1 = *(const float4*)(mrow + p0 + 16 + 8 * half);
            float4 mb1 = *(const float4*)(mrow + p0 + 16 + 8 * half + 4);
            float m0[8] = {ma0.x, ma0.y, ma0.z, ma0.w, mb0.x, mb0.y, mb0.z, mb0.w};
            float m1[8] = {ma1.x, ma1.y, ma1.z, ma1.w, mb1.x, mb1.y, mb1.z, mb1.w};

            float tmax = -1e30f;
#pragma unroll
            for (int r = 0; r < 8; ++r) {
                s0.f[r] = s0.f[r] * inv_sqrt * m0[r];
                s1.f[r] = s1.f[r] * inv_sqrt * m1[r];
                tmax = fmaxf(tmax, fmaxf(s0.f[r], s1.f[r]));
            }
            tmax = fmaxf(tmax, __shfl_xor(tmax, 16));
            float mnew = fmaxf(mrun, tmax);
            float corr = __expf(mrun - mnew);

            float e0[8], e1[8];
            float psum = 0.0f;
#pragma unroll
            for (int r = 0; r < 8; ++r) {
                e0[r] = __expf(s0.f[r] - mnew);
                e1[r] = __expf(s1.f[r] - mnew);
                psum += e0[r] + e1[r];
            }
            psum += __shfl_xor(psum, 16);
            lsum = lsum * corr + psum;
#pragma unroll
            for (int r = 0; r < 8; ++r) { acc0.f[r] *= corr; acc1.f[r] *= corr; }

            // Repack exp'd C-tiles into the B operand (in-lane, no shuffles):
            // B reg v <- tile (v/4), regs 2*(v&3), 2*(v&3)+1, same lane.
            FragB pfrag;
#pragma unroll
            for (int v = 0; v < 8; ++v) {
                const float* src = (v < 4) ? e0 : e1;
                int rr = (v & 3) << 1;
                pfrag.u[v] = pack_bf16(src[rr], src[rr + 1]);
            }

            v16bf v0 = load_frag(sVT + p0, SVT_LD);                       // d rows 0..15
            v16bf v1 = load_frag(sVT + (size_t)16 * SVT_LD + p0, SVT_LD); // d rows 16..31
            acc0.v = wmma32(v0, pfrag.v, acc0.v);
            acc1.v = wmma32(v1, pfrag.v, acc1.v);
            mrun = mnew;
        }

        float inv = 1.0f / lsum;
        // O^T C-layout: (d = 16*t + r + 8*half, g = col); store o[b][g][d] bf16
        uint16_t* orow = Ob + ((size_t)(b * NG) + g) * ND + h * DH + 8 * half;
#pragma unroll
        for (int r = 0; r < 8; r += 2) {
            *(uint32_t*)(orow + r)      = pack_bf16(acc0.f[r] * inv, acc0.f[r + 1] * inv);
            *(uint32_t*)(orow + 16 + r) = pack_bf16(acc1.f[r] * inv, acc1.f[r + 1] * inv);
        }
    }
}

// ---------------------------------------------------------------------------
// Mean over G (vectorized: each thread owns 8 contiguous d columns), then
// final GEMM (16 x 128) @ (128 x 512) + b_out -> f32 out
// ---------------------------------------------------------------------------
__global__ void mean_kernel(const uint16_t* __restrict__ X, uint16_t* __restrict__ M) {
    int i = blockIdx.x * blockDim.x + threadIdx.x;
    if (i >= NB * (ND / 8)) return;            // 256 threads: (b, d8)
    int b = i >> 4, d8 = (i & 15) * 8;
    const uint16_t* p = X + (size_t)b * NG * ND + d8;
    float s[8];
#pragma unroll
    for (int j = 0; j < 8; ++j) s[j] = 0.0f;
    for (int gg = 0; gg < NG; ++gg) {
        uint4 q = *(const uint4*)(p + (size_t)gg * ND);
        uint32_t d[4] = {q.x, q.y, q.z, q.w};
#pragma unroll
        for (int j = 0; j < 4; ++j) {
            s[2 * j]     += bf2f((uint16_t)(d[j] & 0xFFFFu));
            s[2 * j + 1] += bf2f((uint16_t)(d[j] >> 16));
        }
    }
    const float inv = 1.0f / (float)NG;
    uint16_t* dst = M + (size_t)b * ND + d8;
#pragma unroll
    for (int j = 0; j < 4; ++j)
        *(uint32_t*)(dst + 2 * j) = pack_bf16(s[2 * j] * inv, s[2 * j + 1] * inv);
}

__global__ void final_kernel(const uint16_t* __restrict__ Mb, const uint16_t* __restrict__ WoutT,
                             const float* __restrict__ bout, float* __restrict__ out) {
    int job = (int)((blockIdx.x * blockDim.x + threadIdx.x) >> 5);
    if (job >= 32) return;
    int lane = threadIdx.x & 31;
    int half = lane >> 4, col = lane & 15;
    int nt = job;
    // Computed transposed (A = Wout^T tile, B = mean tile): C m = out-col on
    // regs, n = batch on lanes -> f32 epilogue packs into two b128 stores.
    AccF acc;
    float4 b0 = *(const float4*)(bout + nt * 16 + 8 * half);
    float4 b1 = *(const float4*)(bout + nt * 16 + 8 * half + 4);
    acc.f[0] = b0.x; acc.f[1] = b0.y; acc.f[2] = b0.z; acc.f[3] = b0.w;
    acc.f[4] = b1.x; acc.f[5] = b1.y; acc.f[6] = b1.z; acc.f[7] = b1.w;
#pragma unroll
    for (int kc = 0; kc < 4; ++kc) {
        v16bf a = load_frag(WoutT + (size_t)nt * 16 * ND + kc * 32, ND);
        v16bf b = load_frag(Mb + kc * 32, ND);
        acc.v = wmma32(a, b, acc.v);
    }
    float* orow = out + (size_t)col * NP + nt * 16 + 8 * half;
    *(float4*)(orow)     = make_float4(acc.f[0], acc.f[1], acc.f[2], acc.f[3]);
    *(float4*)(orow + 4) = make_float4(acc.f[4], acc.f[5], acc.f[6], acc.f[7]);
}

// ---------------------------------------------------------------------------
// Workspace layout (bytes)
// ---------------------------------------------------------------------------
static constexpr size_t SZ_MASK = (size_t)NG * NP * 4;           // 4,096,000
static constexpr size_t SZ_X    = (size_t)NB * NG * ND * 2;      // 8,192,000
static constexpr size_t SZ_KX   = (size_t)NB * NP * ND * 2;      // 2,097,152
static constexpr size_t SZ_W    = (size_t)2 * ND * ND * 2;       // 65,536
static constexpr size_t SZ_WOUT = (size_t)NP * ND * 2;           // 131,072

static constexpr size_t OFF_MASK  = 0;
static constexpr size_t OFF_XA    = OFF_MASK  + SZ_MASK;
static constexpr size_t OFF_XB    = OFF_XA    + SZ_X;
static constexpr size_t OFF_KX    = OFF_XB    + SZ_X;
static constexpr size_t OFF_Q     = OFF_KX    + SZ_KX;
static constexpr size_t OFF_K     = OFF_Q     + SZ_X;
static constexpr size_t OFF_VT    = OFF_K     + SZ_KX;
static constexpr size_t OFF_O     = OFF_VT    + SZ_KX;
static constexpr size_t OFF_WQT   = OFF_O     + SZ_X;
static constexpr size_t OFF_WKT   = OFF_WQT   + SZ_W;
static constexpr size_t OFF_WVT   = OFF_WKT   + SZ_W;
static constexpr size_t OFF_WOT   = OFF_WVT   + SZ_W;
static constexpr size_t OFF_WOUTT = OFF_WOT   + SZ_W;
static constexpr size_t OFF_MEAN  = OFF_WOUTT + SZ_WOUT;

extern "C" void kernel_launch(void* const* d_in, const int* in_sizes, int n_in,
                              void* d_out, int out_size, void* d_ws, size_t ws_size,
                              hipStream_t stream) {
    (void)in_sizes; (void)n_in; (void)out_size; (void)ws_size;
    const int*   gi    = (const int*)d_in[0];
    const float* expr  = (const float*)d_in[1];
    const int*   pi    = (const int*)d_in[2];
    const float* Wsoft = (const float*)d_in[3];
    const float* gemb  = (const float*)d_in[4];
    const float* pemb  = (const float*)d_in[5];
    const float* Wq    = (const float*)d_in[6];
    const float* bq    = (const float*)d_in[7];
    const float* Wk    = (const float*)d_in[8];
    const float* bk    = (const float*)d_in[9];
    const float* Wv    = (const float*)d_in[10];
    const float* bv    = (const float*)d_in[11];
    const float* Wo    = (const float*)d_in[12];
    const float* bo    = (const float*)d_in[13];
    const float* Wout  = (const float*)d_in[14];
    const float* bout  = (const float*)d_in[15];

    char* ws = (char*)d_ws;
    float*    mask  = (float*)(ws + OFF_MASK);
    uint16_t* xA    = (uint16_t*)(ws + OFF_XA);
    uint16_t* xB    = (uint16_t*)(ws + OFF_XB);
    uint16_t* kxb   = (uint16_t*)(ws + OFF_KX);
    uint16_t* Qb    = (uint16_t*)(ws + OFF_Q);
    uint16_t* Kb    = (uint16_t*)(ws + OFF_K);
    uint16_t* VTb   = (uint16_t*)(ws + OFF_VT);
    uint16_t* Ob    = (uint16_t*)(ws + OFF_O);
    uint16_t* WqT   = (uint16_t*)(ws + OFF_WQT);
    uint16_t* WkT   = (uint16_t*)(ws + OFF_WKT);
    uint16_t* WvT   = (uint16_t*)(ws + OFF_WVT);
    uint16_t* WoT   = (uint16_t*)(ws + OFF_WOT);
    uint16_t* WoutT = (uint16_t*)(ws + OFF_WOUTT);
    uint16_t* Mb    = (uint16_t*)(ws + OFF_MEAN);

    const int TB = 256;
    // Stage 0
    mask_kernel<<<(NG * NP + TB - 1) / TB, TB, 0, stream>>>(gi, Wsoft, mask);
    xinit_kernel<<<(NB * NG * ND + TB - 1) / TB, TB, 0, stream>>>(gi, expr, gemb, xA);
    kxinit_kernel<<<(NB * NP * ND + TB - 1) / TB, TB, 0, stream>>>(pi, pemb, kxb);
    for (int l = 0; l < 2; ++l) {
        size_t o = (size_t)l * ND * ND;
        transpose_bf16_kernel<<<64, TB, 0, stream>>>(Wq + o, WqT + o, ND, ND);
        transpose_bf16_kernel<<<64, TB, 0, stream>>>(Wk + o, WkT + o, ND, ND);
        transpose_bf16_kernel<<<64, TB, 0, stream>>>(Wv + o, WvT + o, ND, ND);
        transpose_bf16_kernel<<<64, TB, 0, stream>>>(Wo + o, WoT + o, ND, ND);
    }
    transpose_bf16_kernel<<<(ND * NP + TB - 1) / TB, TB, 0, stream>>>(Wout, WoutT, ND, NP);

    // Layers
    const uint16_t* xin = xA;
    uint16_t* xout = xB;
    for (int l = 0; l < 2; ++l) {
        size_t wo = (size_t)l * ND * ND;
        size_t bo_off = (size_t)l * ND;
        // Q: 2000 row tiles * 8 col tiles = 16000 waves -> 2000 blocks
        proj_kernel<<<2000, TB, 0, stream>>>(xin, WqT + wo, bq + bo_off, Qb, NB * NG / 16, 0);
        // K,V: 512 row tiles * 8 = 4096 waves -> 512 blocks
        proj_kernel<<<512, TB, 0, stream>>>(kxb, WkT + wo, bk + bo_off, Kb, NB * NP / 16, 0);
        proj_kernel<<<512, TB, 0, stream>>>(kxb, WvT + wo, bv + bo_off, VTb, NB * NP / 16, 1);
        // attention: 64 (b,h) pairs x 8 query-tile chunks, 8 waves/block
        attn_kernel<<<512, TB, ATTN_LDS_BYTES, stream>>>(Qb, Kb, VTb, mask, Ob);
        // output projection
        proj_kernel<<<2000, TB, 0, stream>>>(Ob, WoT + wo, bo + bo_off, xout, NB * NG / 16, 0);
        const uint16_t* t = xin; xin = xout; xout = (uint16_t*)t;
    }

    // Mean over G + final GEMM
    mean_kernel<<<1, TB, 0, stream>>>(xin, Mb);
    final_kernel<<<4, TB, 0, stream>>>(Mb, WoutT, bout, (float*)d_out);
}
